// Weighting_old_52364241273285
// MI455X (gfx1250) — matrix operand order repaired and checked by reading, not compiled, not run on hardware
//
#include <hip/hip_runtime.h>

#define HWPX 4096   // H*W
#define CIN  512    // C
#define DQK  64     // C/8
#define NBATCH 4

typedef __attribute__((ext_vector_type(16))) _Float16     v16h;
typedef __attribute__((ext_vector_type(8)))  float        v8f;
typedef __attribute__((ext_vector_type(4)))  unsigned int v4u;
typedef __attribute__((ext_vector_type(4)))  float        v4f;

union Frag16 { v16h v; v4u q[2]; };

// ---- gfx1250 async global<->LDS copies (ASYNCcnt path), with sync fallback ----
#if defined(__HIP_DEVICE_COMPILE__) && defined(__gfx1250__)
#if __has_builtin(__builtin_amdgcn_global_load_async_to_lds_b128)
#define HAS_ASYNC_LD 1
#endif
#if __has_builtin(__builtin_amdgcn_global_store_async_from_lds_b128)
#define HAS_ASYNC_ST 1
#endif
#endif

// exact parameter type per clang diagnostic: '__attribute__((vector_size(16))) int *'
typedef int v4i_async __attribute__((vector_size(16)));
typedef __attribute__((address_space(1))) v4i_async* gasync_p;
typedef __attribute__((address_space(3))) v4i_async* lasync_p;

__device__ __forceinline__ void async_copy16(const _Float16* g, _Float16* l) {
#if defined(HAS_ASYNC_LD)
  // AS(3) ptr is 32-bit; flat LDS address low 32 bits == LDS offset (ISA 10.2 aperture rules)
  __builtin_amdgcn_global_load_async_to_lds_b128(
      (gasync_p)(unsigned long long)(size_t)g,
      (lasync_p)(unsigned int)(size_t)l, 0, 0);
#else
  *reinterpret_cast<v4u*>(l) = *reinterpret_cast<const v4u*>(g);
#endif
}
__device__ __forceinline__ void async_store16(_Float16* g, const _Float16* l) {
#if defined(HAS_ASYNC_ST)
  __builtin_amdgcn_global_store_async_from_lds_b128(
      (gasync_p)(unsigned long long)(size_t)g,
      (lasync_p)(unsigned int)(size_t)l, 0, 0);
#else
  *reinterpret_cast<v4u*>(g) = *reinterpret_cast<const v4u*>(l);
#endif
}
__device__ __forceinline__ void async_wait0() {
#if defined(HAS_ASYNC_LD) || defined(HAS_ASYNC_ST)
#if __has_builtin(__builtin_amdgcn_s_wait_asynccnt)
  __builtin_amdgcn_s_wait_asynccnt(0);
#else
  asm volatile("s_wait_asynccnt 0x0" ::: "memory");
#endif
#endif
}

// ---- CDNA5 WMMA fragment loaders (layouts per CDNA5 ISA §7.12.2) ----
__device__ __forceinline__ v16h load_a_frag(const _Float16* row_base, int hh) {
  Frag16 f;
  f.q[0] = *reinterpret_cast<const v4u*>(row_base + 8 * hh);
  f.q[1] = *reinterpret_cast<const v4u*>(row_base + 16 + 8 * hh);
  return f.v;
}
__device__ __forceinline__ v16h load_b_frag(const _Float16* row_base, int hh) {
  Frag16 f;
  f.q[0] = *reinterpret_cast<const v4u*>(row_base + 8 * hh);
  f.q[1] = *reinterpret_cast<const v4u*>(row_base + 16 + 8 * hh);
  return f.v;
}
__device__ __forceinline__ v8f wmma16(v16h a, v16h b, v8f c) {
  return __builtin_amdgcn_wmma_f32_16x16x32_f16(false, a, false, b, (short)0, c, false, false);
}

// ============================================================================
// 1x1-conv projection:  Out[i,o] = sum_c X[b,c,i] * W[o,c] + bias[o]  (f16 out)
// OUT_NM==0: Out laid out [HW, Nout];  OUT_NM==1: [Nout, HW]
// (f32->f16 conversion happens while staging, so this one keeps VALU staging)
// ============================================================================
template <int OUT_NM>
__global__ __launch_bounds__(128) void proj_kernel(
    const float* __restrict__ X, const float* __restrict__ Wm,
    const float* __restrict__ bias, _Float16* __restrict__ Out, int Nout) {
  __shared__ _Float16 Al[64 * 40];
  __shared__ _Float16 Bl[64 * 40];
  const int t = threadIdx.x, lane = t & 31, wid = t >> 5;
  const int m0 = blockIdx.x * 64, n0 = blockIdx.y * 64, b = blockIdx.z;
  const float* Xb = X + (size_t)b * CIN * HWPX;
  _Float16* Ob = Out + (size_t)b * HWPX * Nout;

  const int wm = (wid >> 1) * 32, wn = (wid & 1) * 32;
  const int hh = lane >> 4, ln = lane & 15;
  v8f zero = {};
  v8f acc[2][2];
  for (int i = 0; i < 2; ++i)
    for (int j = 0; j < 2; ++j) acc[i][j] = zero;

  for (int c0 = 0; c0 < CIN; c0 += 32) {
    __syncthreads();
#pragma unroll
    for (int i = 0; i < 16; ++i) {
      int idx = t + i * 128;
      int m = idx & 63, k = idx >> 6;
      Al[m * 40 + k] = (_Float16)Xb[(size_t)(c0 + k) * HWPX + m0 + m];
    }
#pragma unroll
    for (int i = 0; i < 16; ++i) {
      int idx = t + i * 128;
      int k = idx & 31, n = idx >> 5;
      Bl[n * 40 + k] = (_Float16)Wm[(size_t)(n0 + n) * CIN + c0 + k];
    }
    __syncthreads();
#pragma unroll
    for (int mf = 0; mf < 2; ++mf) {
      v16h a = load_a_frag(Al + (wm + mf * 16 + ln) * 40, hh);
#pragma unroll
      for (int nf = 0; nf < 2; ++nf) {
        v16h bf = load_b_frag(Bl + (wn + nf * 16 + ln) * 40, hh);
        acc[mf][nf] = wmma16(a, bf, acc[mf][nf]);
      }
    }
  }
#pragma unroll
  for (int mf = 0; mf < 2; ++mf)
#pragma unroll
    for (int nf = 0; nf < 2; ++nf) {
      int n = n0 + wn + nf * 16 + ln;
      float bv = bias[n];
#pragma unroll
      for (int v = 0; v < 8; ++v) {
        int m = m0 + wm + mf * 16 + v + 8 * hh;
        float val = acc[mf][nf][v] + bv;
        if (OUT_NM == 0) Ob[(size_t)m * Nout + n] = (_Float16)val;
        else             Ob[(size_t)n * HWPX + m] = (_Float16)val;
      }
    }
}

// ============================================================================
// Softmax: S = Qt*Kt^T (WMMA), online row max/sum (sweep 1), recompute and
// write normalized P AND P^T via LDS transpose tile (sweep 2), all coalesced.
// Qt,Kt: [B][HW][64] f16.  P,Pt: [B][HW][HW] f16.  Block = 64 rows, 128 threads.
// ============================================================================
__global__ __launch_bounds__(128) void softmax_kernel(
    const _Float16* __restrict__ Qt, const _Float16* __restrict__ Kt,
    _Float16* __restrict__ P, _Float16* __restrict__ Pt) {
  __shared__ _Float16 Ql[64 * 72];
  __shared__ _Float16 Bl[2][128 * 72];
  __shared__ _Float16 Pl[64 * 136];
  const int t = threadIdx.x, lane = t & 31, wid = t >> 5;
  const int i0 = blockIdx.x * 64, b = blockIdx.y;
  const _Float16* Qb = Qt + (size_t)b * HWPX * DQK;
  const _Float16* Kb = Kt + (size_t)b * HWPX * DQK;
  _Float16* Pb = P + (size_t)b * HWPX * HWPX;
  _Float16* Ptb = Pt + (size_t)b * HWPX * HWPX;
  const int hh = lane >> 4, ln = lane & 15;

  // stage Q block (64 rows x 128B), pure row copy -> async
#pragma unroll
  for (int i = 0; i < 4; ++i) {
    int idx = t + i * 128;
    int c8 = idx & 7, m = idx >> 3;
    async_copy16(Qb + (size_t)(i0 + m) * DQK + 8 * c8, &Ql[m * 72 + 8 * c8]);
  }
  auto stageB = [&](int buf, int j0) {
#pragma unroll
    for (int i = 0; i < 8; ++i) {
      int idx = t + i * 128;
      int c8 = idx & 7, n = idx >> 3;
      async_copy16(Kb + (size_t)(j0 + n) * DQK + 8 * c8, &Bl[buf][n * 72 + 8 * c8]);
    }
  };
  stageB(0, 0);
  async_wait0();
  __syncthreads();

  const int mrow = wid * 16 + ln;
  v16h aQ[2];
  aQ[0] = load_a_frag(Ql + mrow * 72 + 0, hh);
  aQ[1] = load_a_frag(Ql + mrow * 72 + 32, hh);

  float rmax[8], rsum[8];
#pragma unroll
  for (int v = 0; v < 8; ++v) { rmax[v] = -3.0e38f; rsum[v] = 0.0f; }

  // ---- sweep 1: statistics (double-buffered) ----
  int buf = 0;
  for (int j0 = 0; j0 < HWPX; j0 += 128) {
    if (j0 + 128 < HWPX) stageB(buf ^ 1, j0 + 128);
    v8f s[8];
#pragma unroll
    for (int nf = 0; nf < 8; ++nf) {
      const _Float16* br = Bl[buf] + (nf * 16 + ln) * 72;
      s[nf] = {};
      s[nf] = wmma16(aQ[0], load_b_frag(br + 0, hh), s[nf]);
      s[nf] = wmma16(aQ[1], load_b_frag(br + 32, hh), s[nf]);
    }
#pragma unroll
    for (int v = 0; v < 8; ++v) {
      float tm = s[0][v];
#pragma unroll
      for (int nf = 1; nf < 8; ++nf) tm = fmaxf(tm, s[nf][v]);
      for (int off = 1; off < 16; off <<= 1) tm = fmaxf(tm, __shfl_xor(tm, off, 32));
      float nm = fmaxf(rmax[v], tm);
      float ts = 0.0f;
#pragma unroll
      for (int nf = 0; nf < 8; ++nf) ts += __expf(s[nf][v] - nm);
      for (int off = 1; off < 16; off <<= 1) ts += __shfl_xor(ts, off, 32);
      rsum[v] = rsum[v] * __expf(rmax[v] - nm) + ts;
      rmax[v] = nm;
    }
    async_wait0();
    __syncthreads();
    buf ^= 1;
  }
  float rinv[8];
#pragma unroll
  for (int v = 0; v < 8; ++v) rinv[v] = 1.0f / rsum[v];

  // ---- sweep 2: recompute, write normalized P (rows) and Pt (cols) ----
  stageB(buf, 0);
  async_wait0();
  __syncthreads();
  for (int j0 = 0; j0 < HWPX; j0 += 128) {
    if (j0 + 128 < HWPX) stageB(buf ^ 1, j0 + 128);
    v8f s[8];
#pragma unroll
    for (int nf = 0; nf < 8; ++nf) {
      const _Float16* br = Bl[buf] + (nf * 16 + ln) * 72;
      s[nf] = {};
      s[nf] = wmma16(aQ[0], load_b_frag(br + 0, hh), s[nf]);
      s[nf] = wmma16(aQ[1], load_b_frag(br + 32, hh), s[nf]);
    }
    __syncthreads();  // Pl from previous iteration fully drained
#pragma unroll
    for (int nf = 0; nf < 8; ++nf) {
      int nloc = nf * 16 + ln;
#pragma unroll
      for (int v = 0; v < 8; ++v) {
        int mloc = wid * 16 + v + 8 * hh;
        Pl[mloc * 136 + nloc] = (_Float16)(__expf(s[nf][v] - rmax[v]) * rinv[v]);
      }
    }
    __syncthreads();
    // P rows: 64 rows x 64 halfs, b128-coalesced (async store path)
#pragma unroll
    for (int i = 0; i < 4; ++i) {
      int idx = t + i * 128;
      int c8 = idx & 7, m = idx >> 3;
      async_store16(Pb + (size_t)(i0 + m) * HWPX + j0 + 8 * c8, &Pl[m * 136 + 8 * c8]);
    }
    // Pt rows: 128 rows x 32 halfs, packed from LDS columns
#pragma unroll
    for (int i = 0; i < 4; ++i) {
      int idx = t + i * 128;
      int cX = idx & 3, n = idx >> 2;
      union { v4u q; _Float16 h[8]; } pk;
#pragma unroll
      for (int e = 0; e < 8; ++e) pk.h[e] = Pl[(8 * cX + e) * 136 + n];
      *reinterpret_cast<v4u*>(Ptb + (size_t)(j0 + n) * HWPX + i0 + 8 * cX) = pk.q;
    }
    async_wait0();
    __syncthreads();
    buf ^= 1;
  }
}

// ============================================================================
// Output GEMM: Out[b,c,i] = gamma * sum_j A[i,j] * Vd[c,j] + Res[b,c,i]
// A: [HW][HW] f16 row-major (P for out1, Pt for out2).  Vd: [C][HW] f16.
// 128x128 tile, 256 threads / 8 waves (64x32 each), async double-buffered LDS.
// ============================================================================
__global__ __launch_bounds__(256) void out_gemm_kernel(
    const _Float16* __restrict__ A, const _Float16* __restrict__ Vd,
    const float* __restrict__ Res, const float* __restrict__ gamma,
    float* __restrict__ Out) {
  __shared__ _Float16 Al[2][128 * 40];
  __shared__ _Float16 Bl[2][128 * 40];
  const int t = threadIdx.x, lane = t & 31, wid = t >> 5;
  const int m0 = blockIdx.x * 128, n0 = blockIdx.y * 128, b = blockIdx.z;
  const _Float16* Ab = A + (size_t)b * HWPX * HWPX;
  const _Float16* Vb = Vd + (size_t)b * CIN * HWPX;
  const float* Rb = Res + (size_t)b * CIN * HWPX;
  float* Ob = Out + (size_t)b * CIN * HWPX;
  const int hh = lane >> 4, ln = lane & 15;
  const int wm = (wid & 1) * 64, wn = (wid >> 1) * 32;

  auto stage = [&](int buf, int k0) {
#pragma unroll
    for (int i = 0; i < 2; ++i) {      // A: 128 rows x 4 chunks
      int idx = t + i * 256;
      int c4 = idx & 3, m = idx >> 2;
      async_copy16(Ab + (size_t)(m0 + m) * HWPX + k0 + 8 * c4, &Al[buf][m * 40 + 8 * c4]);
    }
#pragma unroll
    for (int i = 0; i < 2; ++i) {      // B: 128 chan-rows x 4 chunks
      int idx = t + i * 256;
      int c4 = idx & 3, n = idx >> 2;
      async_copy16(Vb + (size_t)(n0 + n) * HWPX + k0 + 8 * c4, &Bl[buf][n * 40 + 8 * c4]);
    }
  };

  v8f zero = {};
  v8f acc[4][2];
#pragma unroll
  for (int i = 0; i < 4; ++i)
#pragma unroll
    for (int j = 0; j < 2; ++j) acc[i][j] = zero;

  stage(0, 0);
  async_wait0();
  __syncthreads();
  int buf = 0;
  for (int k0 = 0; k0 < HWPX; k0 += 32) {
    if (k0 + 32 < HWPX) stage(buf ^ 1, k0 + 32);   // prefetch overlaps WMMA
    v16h bfr[2];
#pragma unroll
    for (int nf = 0; nf < 2; ++nf)
      bfr[nf] = load_b_frag(Bl[buf] + (wn + nf * 16 + ln) * 40, hh);
#pragma unroll
    for (int mf = 0; mf < 4; ++mf) {
      v16h a = load_a_frag(Al[buf] + (wm + mf * 16 + ln) * 40, hh);
#pragma unroll
      for (int nf = 0; nf < 2; ++nf)
        acc[mf][nf] = wmma16(a, bfr[nf], acc[mf][nf]);
    }
    async_wait0();
    __syncthreads();
    buf ^= 1;
  }

  const float g = gamma[0];
#pragma unroll
  for (int mf = 0; mf < 4; ++mf)
#pragma unroll
    for (int nf = 0; nf < 2; ++nf) {
      int n = n0 + wn + nf * 16 + ln;          // channel c
      int mb = m0 + wm + mf * 16 + 8 * hh;     // 8 consecutive pixels per lane
      const v4f* rp = reinterpret_cast<const v4f*>(Rb + (size_t)n * HWPX + mb);
      v4f* op = reinterpret_cast<v4f*>(Ob + (size_t)n * HWPX + mb);
      v4f r0 = rp[0], r1 = rp[1], o0, o1;
#pragma unroll
      for (int v = 0; v < 4; ++v) o0[v] = g * acc[mf][nf][v] + r0[v];
#pragma unroll
      for (int v = 0; v < 4; ++v) o1[v] = g * acc[mf][nf][v + 4] + r1[v];
      op[0] = o0;
      op[1] = o1;
    }
}

// ============================================================================
extern "C" void kernel_launch(void* const* d_in, const int* in_sizes, int n_in,
                              void* d_out, int out_size, void* d_ws, size_t ws_size,
                              hipStream_t stream) {
  (void)in_sizes; (void)n_in; (void)out_size; (void)ws_size;
  const float* x   = (const float*)d_in[0];
  const float* y   = (const float*)d_in[1];
  const float* wq  = (const float*)d_in[2];
  const float* bq  = (const float*)d_in[3];
  const float* wk  = (const float*)d_in[4];
  const float* bk  = (const float*)d_in[5];
  const float* wv1 = (const float*)d_in[6];
  const float* bv1 = (const float*)d_in[7];
  const float* wv2 = (const float*)d_in[8];
  const float* bv2 = (const float*)d_in[9];
  const float* g1  = (const float*)d_in[10];
  const float* g2  = (const float*)d_in[11];
  float* out1 = (float*)d_out;
  float* out2 = out1 + (size_t)NBATCH * CIN * HWPX;

  // workspace (f16): Qt 2MB + Kt 2MB + V1d/V2d 16.8MB each + P/Pt 134MB each
  char* ws = (char*)d_ws;
  size_t off = 0;
  auto alloc = [&](size_t bytes) {
    off = (off + 255) & ~(size_t)255;
    void* p = ws + off;
    off += bytes;
    return p;
  };
  _Float16* Qt  = (_Float16*)alloc((size_t)NBATCH * HWPX * DQK * 2);
  _Float16* Kt  = (_Float16*)alloc((size_t)NBATCH * HWPX * DQK * 2);
  _Float16* V1d = (_Float16*)alloc((size_t)NBATCH * CIN * HWPX * 2);
  _Float16* V2d = (_Float16*)alloc((size_t)NBATCH * CIN * HWPX * 2);
  _Float16* P   = (_Float16*)alloc((size_t)NBATCH * HWPX * HWPX * 2);
  _Float16* Pt  = (_Float16*)alloc((size_t)NBATCH * HWPX * HWPX * 2);

  proj_kernel<0><<<dim3(HWPX / 64, DQK / 64, NBATCH), 128, 0, stream>>>(x, wq, bq, Qt, DQK);
  proj_kernel<0><<<dim3(HWPX / 64, DQK / 64, NBATCH), 128, 0, stream>>>(y, wk, bk, Kt, DQK);
  proj_kernel<1><<<dim3(HWPX / 64, CIN / 64, NBATCH), 128, 0, stream>>>(x, wv1, bv1, V1d, CIN);
  proj_kernel<1><<<dim3(HWPX / 64, CIN / 64, NBATCH), 128, 0, stream>>>(y, wv2, bv2, V2d, CIN);
  softmax_kernel<<<dim3(HWPX / 64, NBATCH), 128, 0, stream>>>(Qt, Kt, P, Pt);
  out_gemm_kernel<<<dim3(HWPX / 128, CIN / 128, NBATCH), 256, 0, stream>>>(P, V1d, x, g1, out1);
  out_gemm_kernel<<<dim3(HWPX / 128, CIN / 128, NBATCH), 256, 0, stream>>>(Pt, V2d, y, g2, out2);
}
